// GCNNFeatureExtractor_1503238553663
// MI455X (gfx1250) — compile-verified, matrix operand order checked by reading, not run on hardware
//
#include <hip/hip_runtime.h>
#include <hip/hip_bf16.h>

// ---------------------------------------------------------------------------
// CDNA5 (gfx1250) implementation of the relational-GAT reaction model.
//
// Dense math (per-relation GEMMs h @ W[r], MLP head) -> v_wmma_f32_16x16x32_f16
// (wave32 WMMA, f32 accumulate), with A buffers row-padded to 16 so operand
// loads are unconditional global_load_b128 pairs (no exec masking in the hot
// loop) and K templated so the MAC loop fully unrolls.
// Sparse graph ops (segment softmax / scatter) -> flat coalesced kernels with
// f32/u32 global atomics (L2-resident targets; 192 MB L2 holds agg easily).
// ---------------------------------------------------------------------------

typedef __attribute__((ext_vector_type(16))) _Float16 v16h;
typedef __attribute__((ext_vector_type(8)))  float    v8f;

static __device__ __forceinline__ unsigned fenc(float f) {
    unsigned u = __float_as_uint(f);
    return (u & 0x80000000u) ? ~u : (u | 0x80000000u);
}
static __device__ __forceinline__ float fdec(unsigned u) {
    unsigned v = (u & 0x80000000u) ? (u ^ 0x80000000u) : ~u;
    return __uint_as_float(v);
}

// Load one 16x32 f16 operand tile (A or B^T), row-major, row stride = ld.
// Per ISA 7.12.2: lane L holds row row0 + (L&15); half = L>>4 selects the
// K-subblocks: elems 0..7 -> k0+8*half, elems 8..15 -> k0+16+8*half.
// Caller guarantees rows are in bounds (buffers are row-padded to 16).
static __device__ __forceinline__ v16h load_tile(const _Float16* __restrict__ base,
                                                 int row0, int ld, int k0, int lane) {
    const int r    = row0 + (lane & 15);
    const int half = lane >> 4;
    const _Float16* p = base + (size_t)r * ld + k0 + 8 * half;
    v16h t;
    ((uint4*)&t)[0] = *(const uint4*)(p);        // 8 halves, 16B aligned
    ((uint4*)&t)[1] = *(const uint4*)(p + 16);   // next K-subblock
    return t;
}

// Generic f16 WMMA GEMM: C[z] = A @ Bt[z]^T  (A: [Mpad,K] row-major f16,
// Bt: [Ncols,K] row-major f16 == B pre-transposed), C f32 row-major ldc.
// Block = 128 threads = 4 waves; each wave owns one 16x16 output tile.
// Optional epilogue: + bias[col], PReLU(slope). storeNlimit bounds columns
// (handles the 703-wide output padded to 704). K is compile-time -> the
// load/WMMA sequence is fully unrolled straight-line code.
template <int K>
__global__ void wmma_gemm_f16(const _Float16* __restrict__ A,
                              const _Float16* __restrict__ Bt,
                              float* __restrict__ C,
                              int M, int Ncols,
                              size_t strideB, size_t strideC,
                              int ldc, int storeNlimit,
                              const float* __restrict__ bias,
                              const float* __restrict__ prelu) {
    const int lane = threadIdx.x & 31;
    const int wave = threadIdx.x >> 5;
    const int half = lane >> 4;
    const int row0 = blockIdx.x * 16;
    const int col0 = (blockIdx.y * 4 + wave) * 16;
    if (col0 >= Ncols) return;                    // wave-uniform: EXEC stays full
    const _Float16* Bz = Bt + strideB * blockIdx.z;
    float*          Cz = C  + strideC * blockIdx.z;

    v8f acc = {};
#pragma unroll
    for (int k0 = 0; k0 < K; k0 += 32) {
        v16h a = load_tile(A,  row0, K, k0, lane);
        v16h b = load_tile(Bz, col0, K, k0, lane);
        acc = __builtin_amdgcn_wmma_f32_16x16x32_f16(
            /*neg_a=*/false, a, /*neg_b=*/false, b,
            /*c_mod=*/(short)0, acc, /*reuse_a=*/false, /*reuse_b=*/false);
    }

    const int col = col0 + (lane & 15);
    if (col >= storeNlimit) return;
    const float bv = bias ? bias[col] : 0.0f;
#pragma unroll
    for (int j = 0; j < 8; ++j) {
        const int row = row0 + j + 8 * half;      // ISA C/D layout
        if (row < M) {
            float v = acc[j] + bv;
            if (prelu) { const float sl = prelu[0]; v = (v > 0.0f) ? v : sl * v; }
            Cz[(size_t)row * ldc + col] = v;
        }
    }
}

// f32 -> f16 cast with zero padding for i in [nvalid, ntotal).
__global__ void cast_f16_pad_kernel(const float* __restrict__ x, _Float16* __restrict__ y,
                                    size_t nvalid, size_t ntotal) {
    size_t i = (size_t)blockIdx.x * blockDim.x + threadIdx.x;
    if (i < ntotal) y[i] = (i < nvalid) ? (_Float16)x[i] : (_Float16)0.f;
}

// Per relation r: ws[r] = W[r]@a_src[r], wd[r] = W[r]@a_dst[r]  (exact f32),
// and Wt[r] = W[r]^T in f16 for the WMMA B operand. One thread per (r,k).
__global__ void prep_w_kernel(const float* __restrict__ W,
                              const float* __restrict__ a_s,
                              const float* __restrict__ a_d,
                              _Float16* __restrict__ Wt,
                              float* __restrict__ wsv, float* __restrict__ wdv, int R) {
    int idx = blockIdx.x * blockDim.x + threadIdx.x;
    if (idx >= R * 64) return;
    const int r = idx >> 6, k = idx & 63;
    const float* Wr = W + (size_t)r * 4096;
    float s = 0.f, d = 0.f;
    for (int e = 0; e < 64; ++e) {
        const float w = Wr[k * 64 + e];
        s += w * a_s[r * 64 + e];
        d += w * a_d[r * 64 + e];
        Wt[(size_t)r * 4096 + e * 64 + k] = (_Float16)w;   // transpose-cast
    }
    wsv[idx] = s;
    wdv[idx] = d;
}

// att_s[r,n] = h[n] . ws[r] ; att_d[r,n] = h[n] . wd[r]
// (algebraic identity: einsum('rne,re->rn', h@W, a) == h . (W@a))
__global__ void att_kernel(const float* __restrict__ h,
                           const float* __restrict__ wsv, const float* __restrict__ wdv,
                           float* __restrict__ att_s, float* __restrict__ att_d,
                           int N, int R) {
    size_t idx = (size_t)blockIdx.x * blockDim.x + threadIdx.x;
    if (idx >= (size_t)R * N) return;
    const int r = (int)(idx / N), n = (int)(idx % N);
    const float* hr = h + (size_t)n * 64;
    const float* av = wsv + r * 64;
    const float* bv = wdv + r * 64;
    float s = 0.f, d = 0.f;
#pragma unroll 8
    for (int k = 0; k < 64; ++k) { const float hv = hr[k]; s += hv * av[k]; d += hv * bv[k]; }
    att_s[idx] = s;
    att_d[idx] = d;
}

// scores = leaky_relu(att_s[rel,src] + att_d[rel,dst]); segment max via
// order-preserving u32 atomicMax (global_atomic_max_u32).
__global__ void edge_score_kernel(const float* __restrict__ att_s, const float* __restrict__ att_d,
                                  const int* __restrict__ src, const int* __restrict__ dst,
                                  const int* __restrict__ rel,
                                  float* __restrict__ scores, unsigned* __restrict__ m_enc,
                                  int E, int N) {
    int e = blockIdx.x * blockDim.x + threadIdx.x;
    if (e >= E) return;
    const int r = rel[e];
    const int de = dst[e];
    float s = att_s[(size_t)r * N + src[e]] + att_d[(size_t)r * N + de];
    s = (s > 0.f) ? s : 0.2f * s;                 // NEG_SLOPE = 0.2
    scores[e] = s;
    atomicMax(m_enc + de, fenc(s));
}

// ex = exp(score - m[dst]); segment sum of ex.
__global__ void edge_exp_kernel(const float* __restrict__ scores,
                                const int* __restrict__ dst,
                                const unsigned* __restrict__ m_enc,
                                float* __restrict__ ssum, float* __restrict__ exb, int E) {
    int e = blockIdx.x * blockDim.x + threadIdx.x;
    if (e >= E) return;
    const int de = dst[e];
    float m = fdec(m_enc[de]);
    if (!__builtin_isfinite(m)) m = 0.0f;         // matches jnp.where(isfinite(m), m, 0)
    const float ex = __expf(scores[e] - m);
    exb[e] = ex;
    atomicAdd(ssum + de, ex);
}

// agg[dst] += (ex/(ssum+1e-9)) * hW[rel, src]; one thread per (edge, dim).
__global__ void msg_kernel(const float* __restrict__ hW,
                           const float* __restrict__ exb, const float* __restrict__ ssum,
                           const int* __restrict__ src, const int* __restrict__ dst,
                           const int* __restrict__ rel,
                           float* __restrict__ agg, int E, int N) {
    size_t idx = (size_t)blockIdx.x * blockDim.x + threadIdx.x;
    if (idx >= (size_t)E * 64) return;
    const int e = (int)(idx >> 6);
    const int d = (int)(idx & 63);
    const int de = dst[e];
    const float alpha = exb[e] / (ssum[de] + 1e-9f);
    const float v = alpha * hW[(((size_t)rel[e] * N + src[e]) << 6) + d];
    atomicAdd(agg + ((size_t)de << 6) + d, v);
}

// ELU + padded f16 shadow copy for the next WMMA GEMM.
__global__ void elu_cast_kernel(const float* __restrict__ x,
                                float* __restrict__ yf, _Float16* __restrict__ yh,
                                size_t nvalid, size_t ntotal) {
    size_t i = (size_t)blockIdx.x * blockDim.x + threadIdx.x;
    if (i >= ntotal) return;
    if (i < nvalid) {
        const float v = x[i];
        const float y = (v > 0.f) ? v : (__expf(v) - 1.f);
        yf[i] = y;
        yh[i] = (_Float16)y;
    } else {
        yh[i] = (_Float16)0.f;                    // zero the 16-row pad
    }
}

// out[seg[n], colOff + d] += x[n, d]   (d = 0..63)
__global__ void segsum_rows_kernel(const float* __restrict__ x, const int* __restrict__ seg,
                                   float* __restrict__ out, int Nrows, int outLd, int colOff) {
    size_t idx = (size_t)blockIdx.x * blockDim.x + threadIdx.x;
    if (idx >= (size_t)Nrows * 64) return;
    const int n = (int)(idx >> 6);
    const int d = (int)(idx & 63);
    atomicAdd(out + (size_t)seg[n] * outLd + colOff + d, x[idx]);
}

// dst[NcolPad, K] = transpose-cast of src[K, Ncol], zero-padding rows >= Ncol.
__global__ void tcast_kernel(const float* __restrict__ src, _Float16* __restrict__ dst,
                             int K, int Ncol, int NcolPad) {
    size_t idx = (size_t)blockIdx.x * blockDim.x + threadIdx.x;
    if (idx >= (size_t)NcolPad * K) return;
    const int n = (int)(idx / K), k = (int)(idx % K);
    dst[idx] = (n < Ncol) ? (_Float16)src[(size_t)k * Ncol + n] : (_Float16)0.f;
}

static inline unsigned cdiv(size_t a, size_t b) { return (unsigned)((a + b - 1) / b); }

extern "C" void kernel_launch(void* const* d_in, const int* in_sizes, int n_in,
                              void* d_out, int out_size, void* d_ws, size_t ws_size,
                              hipStream_t stream) {
    (void)in_sizes; (void)n_in; (void)out_size; (void)ws_size;
    constexpr int N = 100000, E = 800000, R = 8, D = 64;
    constexpr int NM = 5000, NMP = 5008;          // mol rows padded to 16
    constexpr int ER = 40000, RR = 4, NB = 1024;
    constexpr int RD = 128, DH = 512, DO = 703, DOP = 704;

    const float* node_feats = (const float*)d_in[0];
    const int*   edge_src   = (const int*)d_in[1];
    const int*   edge_dst   = (const int*)d_in[2];
    const int*   edge_rel   = (const int*)d_in[3];
    const int*   node2mol   = (const int*)d_in[4];
    const int*   rxn_src    = (const int*)d_in[5];
    const int*   rxn_dst    = (const int*)d_in[6];
    const int*   rxn_rel    = (const int*)d_in[7];
    const int*   mol2rxn    = (const int*)d_in[8];
    const float* W1     = (const float*)d_in[9];
    const float* a_src1 = (const float*)d_in[10];
    const float* a_dst1 = (const float*)d_in[11];
    const float* W2     = (const float*)d_in[12];
    const float* a_src2 = (const float*)d_in[13];
    const float* a_dst2 = (const float*)d_in[14];
    const float* Wr     = (const float*)d_in[15];
    const float* a_srcr = (const float*)d_in[16];
    const float* a_dstr = (const float*)d_in[17];
    const float* w_fc1  = (const float*)d_in[18];
    const float* b_fc1  = (const float*)d_in[19];
    const float* prelu1 = (const float*)d_in[20];
    const float* w_fc2  = (const float*)d_in[21];
    const float* b_fc2  = (const float*)d_in[22];
    float* out = (float*)d_out;

    // ---- carve workspace (dominant: hW = 204.8 MB) ----
    char* wsb = (char*)d_ws;
    size_t off = 0;
    auto carve = [&](size_t bytes) -> void* {
        void* p = wsb + off;
        off += (bytes + 255) & ~(size_t)255;
        return p;
    };
    float*    hW     = (float*)carve((size_t)R * N * D * 4);
    _Float16* h16    = (_Float16*)carve((size_t)N * D * 2);
    float*    h2     = (float*)carve((size_t)N * D * 4);
    _Float16* h2_16  = (_Float16*)carve((size_t)N * D * 2);
    float*    agg    = (float*)carve((size_t)N * D * 4);
    float*    att_s  = (float*)carve((size_t)R * N * 4);
    float*    att_d  = (float*)carve((size_t)R * N * 4);
    unsigned* m_enc  = (unsigned*)carve((size_t)N * 4);
    float*    ssum   = (float*)carve((size_t)N * 4);
    float*    scores = (float*)carve((size_t)E * 4);
    float*    exb    = (float*)carve((size_t)E * 4);
    _Float16* Wt     = (_Float16*)carve((size_t)R * D * D * 2);
    float*    wsv    = (float*)carve((size_t)R * D * 4);
    float*    wdv    = (float*)carve((size_t)R * D * 4);
    float*    mol    = (float*)carve((size_t)NM * D * 4);
    _Float16* mol16  = (_Float16*)carve((size_t)NMP * D * 2);   // padded rows
    float*    molr   = (float*)carve((size_t)NM * D * 4);
    _Float16* molr16 = (_Float16*)carve((size_t)NMP * D * 2);
    float*    feat   = (float*)carve((size_t)NB * RD * 4);
    _Float16* feat16 = (_Float16*)carve((size_t)NB * RD * 2);
    float*    hm     = (float*)carve((size_t)NB * DH * 4);
    _Float16* hm16   = (_Float16*)carve((size_t)NB * DH * 2);
    _Float16* w1t    = (_Float16*)carve((size_t)DH * RD * 2);
    _Float16* w2t    = (_Float16*)carve((size_t)DOP * DH * 2);

    // K-templated dispatch (K in {64, 128, 512}).
    auto gemm = [&](const _Float16* A, const _Float16* Bt, float* C, int M, int Ncols, int K,
                    size_t sB, size_t sC, int nz, int ldc, int lim,
                    const float* bias, const float* pr) {
        dim3 g(cdiv(M, 16), cdiv(Ncols, 64), nz);
        if (K == 64) {
            wmma_gemm_f16<64><<<g, 128, 0, stream>>>(A, Bt, C, M, Ncols, sB, sC, ldc, lim, bias, pr);
        } else if (K == 128) {
            wmma_gemm_f16<128><<<g, 128, 0, stream>>>(A, Bt, C, M, Ncols, sB, sC, ldc, lim, bias, pr);
        } else {
            wmma_gemm_f16<512><<<g, 128, 0, stream>>>(A, Bt, C, M, Ncols, sB, sC, ldc, lim, bias, pr);
        }
    };

    // One RGAT layer (without the trailing ELU). hh16 must be row-padded to 16.
    auto rgat = [&](const float* hf, const _Float16* hh16, const float* Wf,
                    const float* asv, const float* adv,
                    const int* es, const int* ed, const int* er,
                    int Nn, int Ee, int Rr, float* hWout, float* aggout) {
        prep_w_kernel<<<cdiv((size_t)Rr * 64, 256), 256, 0, stream>>>(Wf, asv, adv, Wt, wsv, wdv, Rr);
        // hW[r] = h @ W[r]  -- the WMMA hotspot (batched over relations via grid.z)
        gemm(hh16, Wt, hWout, Nn, D, D, (size_t)D * D, (size_t)Nn * D, Rr, D, D, nullptr, nullptr);
        att_kernel<<<cdiv((size_t)Rr * Nn, 256), 256, 0, stream>>>(hf, wsv, wdv, att_s, att_d, Nn, Rr);
        hipMemsetAsync(m_enc, 0, (size_t)Nn * 4, stream);
        hipMemsetAsync(ssum, 0, (size_t)Nn * 4, stream);
        hipMemsetAsync(aggout, 0, (size_t)Nn * D * 4, stream);
        edge_score_kernel<<<cdiv(Ee, 256), 256, 0, stream>>>(att_s, att_d, es, ed, er, scores, m_enc, Ee, Nn);
        edge_exp_kernel<<<cdiv(Ee, 256), 256, 0, stream>>>(scores, ed, m_enc, ssum, exb, Ee);
        msg_kernel<<<cdiv((size_t)Ee * 64, 256), 256, 0, stream>>>(hWout, exb, ssum, es, ed, er, aggout, Ee, Nn);
    };

    // ---- atom RGAT layer 1 ----
    cast_f16_pad_kernel<<<cdiv((size_t)N * D, 256), 256, 0, stream>>>(
        node_feats, h16, (size_t)N * D, (size_t)N * D);
    rgat(node_feats, h16, W1, a_src1, a_dst1, edge_src, edge_dst, edge_rel, N, E, R, hW, agg);
    elu_cast_kernel<<<cdiv((size_t)N * D, 256), 256, 0, stream>>>(
        agg, h2, h2_16, (size_t)N * D, (size_t)N * D);

    // ---- atom RGAT layer 2 ----
    rgat(h2, h2_16, W2, a_src2, a_dst2, edge_src, edge_dst, edge_rel, N, E, R, hW, agg);
    elu_cast_kernel<<<cdiv((size_t)N * D, 256), 256, 0, stream>>>(
        agg, h2, h2_16, (size_t)N * D, (size_t)N * D);

    // ---- molecule readout ----
    hipMemsetAsync(mol, 0, (size_t)NM * D * 4, stream);
    segsum_rows_kernel<<<cdiv((size_t)N * D, 256), 256, 0, stream>>>(h2, node2mol, mol, N, D, 0);

    // ---- reaction-graph RGAT over molecules (A padded 5000 -> 5008 rows) ----
    cast_f16_pad_kernel<<<cdiv((size_t)NMP * D, 256), 256, 0, stream>>>(
        mol, mol16, (size_t)NM * D, (size_t)NMP * D);
    rgat(mol, mol16, Wr, a_srcr, a_dstr, rxn_src, rxn_dst, rxn_rel, NM, ER, RR, hW, agg);
    elu_cast_kernel<<<cdiv((size_t)NMP * D, 256), 256, 0, stream>>>(
        agg, molr, molr16, (size_t)NM * D, (size_t)NMP * D);

    // ---- reaction readout: feat = [segsum(molr) | segsum(mol)] ----
    hipMemsetAsync(feat, 0, (size_t)NB * RD * 4, stream);
    segsum_rows_kernel<<<cdiv((size_t)NM * D, 256), 256, 0, stream>>>(molr, mol2rxn, feat, NM, RD, 0);
    segsum_rows_kernel<<<cdiv((size_t)NM * D, 256), 256, 0, stream>>>(mol, mol2rxn, feat, NM, RD, D);

    // ---- MLP head (WMMA): Linear -> PReLU -> Linear ----
    cast_f16_pad_kernel<<<cdiv((size_t)NB * RD, 256), 256, 0, stream>>>(
        feat, feat16, (size_t)NB * RD, (size_t)NB * RD);
    tcast_kernel<<<cdiv((size_t)DH * RD, 256), 256, 0, stream>>>(w_fc1, w1t, RD, DH, DH);
    gemm(feat16, w1t, hm, NB, DH, RD, 0, 0, 1, DH, DH, b_fc1, prelu1);

    cast_f16_pad_kernel<<<cdiv((size_t)NB * DH, 256), 256, 0, stream>>>(
        hm, hm16, (size_t)NB * DH, (size_t)NB * DH);
    tcast_kernel<<<cdiv((size_t)DOP * DH, 256), 256, 0, stream>>>(w_fc2, w2t, DH, DO, DOP);
    gemm(hm16, w2t, out, NB, DOP, DH, 0, 0, 1, DO, DO, b_fc2, nullptr);
}
